// Sub_Clust_Cos_GLM_4286377361600
// MI455X (gfx1250) — matrix operand order repaired and checked by reading, not compiled, not run on hardware
//
#include <hip/hip_runtime.h>

typedef __attribute__((ext_vector_type(16))) _Float16 v16h;
typedef __attribute__((ext_vector_type(8)))  float    v8f;

#define B_     8
#define TDATA  5000
#define ENO    2000
#define INO    200
#define SUB    20
#define HID    16
#define TNO    400
#define NBASIS 24
#define KPADE  2016   // 63*32 (E reduction padded)
#define KPADI  224    // 7*32  (I reduction padded)
#define TKPAD  416    // 13*32 (conv tap dim padded)
#define NKSTEP 13
#define LEFTPAD 416
#define TPAD   5600   // LEFTPAD + 5000 + right slack
#define NCHUNK 40     // 40*128 >= 5000

// workspace offsets (bytes)
#define OFF_CT_E   0u
#define OFF_CT_I   129024u     // 2016*32*2
#define OFF_FRAG_E 143360u
#define OFF_FRAG_I 409600u     // +20*13*32*16*2
#define OFF_SYN_E  675840u
#define OFF_SYN_I  2467840u    // +8*20*5600*2
#define WS_BYTES   4259840u

union FragH {
  v16h v;
  _Float16 h[16];
  unsigned int u[8];
  uint4 q[2];
};

// ---- CDNA5 async global->LDS copy (ASYNCcnt path), with safe fallback ----
typedef __attribute__((address_space(1))) int gas_int;
typedef __attribute__((address_space(3))) int las_int;

__device__ inline void async_copy_b32(const unsigned int* gsrc, unsigned int* lds_dst) {
#if defined(__gfx1250__) && __has_builtin(__builtin_amdgcn_global_load_async_to_lds_b32)
  __builtin_amdgcn_global_load_async_to_lds_b32((gas_int*)gsrc, (las_int*)lds_dst, 0, 0);
#else
  *lds_dst = *gsrc;
#endif
}
__device__ inline void wait_async() {
#if defined(__gfx1250__) && __has_builtin(__builtin_amdgcn_global_load_async_to_lds_b32)
#if __has_builtin(__builtin_amdgcn_s_wait_asynccnt)
  __builtin_amdgcn_s_wait_asynccnt(0);
#else
  asm volatile("s_wait_asynccnt 0" ::: "memory");
#endif
#endif
}

// ---------------- k0: zero workspace (pads must be 0) ----------------
__global__ void k0_zero(unsigned int* ws, unsigned int n) {
  unsigned int i = blockIdx.x * blockDim.x + threadIdx.x;
  unsigned int stride = gridDim.x * blockDim.x;
  for (; i < n; i += stride) ws[i] = 0u;
}

// ------- k1: column softmax over sub dim; also scaled f16 transpose -------
__global__ void k1_softmax(const float* __restrict__ CeR, const float* __restrict__ CiR,
                           const float* __restrict__ Escale, const float* __restrict__ Iscale,
                           const float* __restrict__ temp,
                           float* __restrict__ oCe, float* __restrict__ oCi,
                           _Float16* __restrict__ Ct_e, _Float16* __restrict__ Ct_i) {
  int idx = blockIdx.x * blockDim.x + threadIdx.x;
  float tinv = 1.0f / temp[0];
  if (idx < ENO) {
    int e = idx;
    float vals[SUB];
    float m = -1e30f;
    for (int s = 0; s < SUB; s++) { float v = CeR[s * ENO + e] * tinv; vals[s] = v; m = fmaxf(m, v); }
    float sum = 0.f;
    for (int s = 0; s < SUB; s++) { float v = expf(vals[s] - m); vals[s] = v; sum += v; }
    float inv = 1.f / sum;
    float sc = expf(Escale[e]);
    for (int s = 0; s < SUB; s++) {
      float p = vals[s] * inv;
      oCe[s * ENO + e] = p;
      Ct_e[e * 32 + s] = (_Float16)(p * sc);   // pad cols 20..31 & rows >=2000 are zero (k0)
    }
  } else if (idx < ENO + INO) {
    int e = idx - ENO;
    float vals[SUB];
    float m = -1e30f;
    for (int s = 0; s < SUB; s++) { float v = CiR[s * INO + e] * tinv; vals[s] = v; m = fmaxf(m, v); }
    float sum = 0.f;
    for (int s = 0; s < SUB; s++) { float v = expf(vals[s] - m); vals[s] = v; sum += v; }
    float inv = 1.f / sum;
    float sc = expf(Iscale[e]);
    for (int s = 0; s < SUB; s++) {
      float p = vals[s] * inv;
      oCi[s * INO + e] = p;
      Ct_i[e * 32 + s] = (_Float16)(p * sc);
    }
  }
}

// ------- k3: build flipped conv kernels, pre-swizzled into WMMA A layout -------
__global__ void k3_kernels(const float* __restrict__ We1, const float* __restrict__ Wi1,
                           _Float16* __restrict__ frag_e, _Float16* __restrict__ frag_i) {
  int idx = blockIdx.x * blockDim.x + threadIdx.x;
  const int per = SUB * HID * TKPAD;
  if (idx >= 2 * per) return;
  int which = idx / per;
  int r = idx % per;
  int row = r / TKPAD;        // s*16+h
  int kk = r % TKPAD;         // tap index after flip, padded to 416
  const float* W = which ? Wi1 : We1;
  float val = 0.f;
  if (kk < TNO) {
    float x = (float)(TNO - 1 - kk);                // flip: tap kk multiplies x[t-399+kk]
    float raw = 6.0f * logf(x + 1.0f + 1e-7f);
    #pragma unroll 1
    for (int c = 0; c < NBASIS; c++) {
      float d = raw - 1.57079632679f * (float)c;
      float bas = (d >= -3.14159265359f && d <= 3.14159265359f) ? (0.5f * cosf(d) + 0.5f) : 0.f;
      val += W[row * NBASIS + c] * bas;
    }
  }
  // A fragment in this code's contiguous k order (k4's B matches): lane = g*16 + M, elem = blk*8 + id, K = g*16+blk*8+id
  int s = row / HID, h = row % HID;
  int kstep = kk >> 5, rr = kk & 31;
  int id = rr & 7, g = (rr >> 4) & 1, blk = (rr >> 3) & 1;
  _Float16* frag = which ? frag_i : frag_e;
  frag[(((size_t)(s * NKSTEP + kstep)) * 32 + (g * 16 + h)) * 16 + (blk * 8 + id)] = (_Float16)val;
}

// ------- k2: syn GEMM via WMMA: syn[b,s,t] = (S * C_scaled^T), f16 in / f32 acc -------
__global__ void __launch_bounds__(128) k2_syn(const float* __restrict__ Se, const float* __restrict__ Si,
                                              const _Float16* __restrict__ Ct_e, const _Float16* __restrict__ Ct_i,
                                              _Float16* __restrict__ syn_e, _Float16* __restrict__ syn_i) {
  int wave = (blockIdx.x * blockDim.x + threadIdx.x) >> 5;   // 2504 waves = 8 b * 313 M-tiles
  int lane = threadIdx.x & 31;
  if (wave >= B_ * 313) return;
  int b = wave / 313, mt = wave % 313;
  int t0 = mt * 16;
  int g = lane >> 4, m = lane & 15;
  int t = t0 + m; if (t > TDATA - 1) t = TDATA - 1;           // clamp loads; stores guarded
  const float* rowE = Se + ((size_t)b * TDATA + t) * ENO;
  const float* rowI = Si + ((size_t)b * TDATA + t) * INO;

  v8f ae0 = {}, ae1 = {}, ai0 = {}, ai1 = {};
  const float4 z4 = make_float4(0.f, 0.f, 0.f, 0.f);

  for (int ks = 0; ks < KPADE / 32; ks++) {
    int e0 = ks * 32;
    int kb0 = e0 + g * 8, kb1 = kb0 + 16;
    if (kb0 + 32 < ENO) __builtin_prefetch(rowE + kb0 + 32, 0, 1);   // cover next K-step
    float4 f0 = (kb0 < ENO) ? *(const float4*)(rowE + kb0)     : z4;
    float4 f1 = (kb0 < ENO) ? *(const float4*)(rowE + kb0 + 4) : z4;
    float4 f2 = (kb1 < ENO) ? *(const float4*)(rowE + kb1)     : z4;
    float4 f3 = (kb1 < ENO) ? *(const float4*)(rowE + kb1 + 4) : z4;
    FragH A;
    A.h[0]=(_Float16)f0.x; A.h[1]=(_Float16)f0.y; A.h[2]=(_Float16)f0.z; A.h[3]=(_Float16)f0.w;
    A.h[4]=(_Float16)f1.x; A.h[5]=(_Float16)f1.y; A.h[6]=(_Float16)f1.z; A.h[7]=(_Float16)f1.w;
    A.h[8]=(_Float16)f2.x; A.h[9]=(_Float16)f2.y; A.h[10]=(_Float16)f2.z; A.h[11]=(_Float16)f2.w;
    A.h[12]=(_Float16)f3.x; A.h[13]=(_Float16)f3.y; A.h[14]=(_Float16)f3.z; A.h[15]=(_Float16)f3.w;
    // B: N col = lane%16; elem s = K (same table order as A); Ct row stride 32 halfs
    const _Float16* bp = Ct_e + ((size_t)e0) * 32 + m;
    FragH B0, B1;
    for (int q = 0; q < 16; q++) { int kq = (q < 8) ? (g * 8 + q) : (16 + g * 8 + q - 8); B0.h[q] = bp[kq * 32]; B1.h[q] = bp[kq * 32 + 16]; }
    ae0 = __builtin_amdgcn_wmma_f32_16x16x32_f16(false, A.v, false, B0.v, (short)0, ae0, false, false);
    ae1 = __builtin_amdgcn_wmma_f32_16x16x32_f16(false, A.v, false, B1.v, (short)0, ae1, false, false);
  }
  for (int ks = 0; ks < KPADI / 32; ks++) {
    int e0 = ks * 32;
    int kb0 = e0 + g * 8, kb1 = kb0 + 16;
    float4 f0 = (kb0 < INO) ? *(const float4*)(rowI + kb0)     : z4;
    float4 f1 = (kb0 < INO) ? *(const float4*)(rowI + kb0 + 4) : z4;
    float4 f2 = (kb1 < INO) ? *(const float4*)(rowI + kb1)     : z4;
    float4 f3 = (kb1 < INO) ? *(const float4*)(rowI + kb1 + 4) : z4;
    FragH A;
    A.h[0]=(_Float16)f0.x; A.h[1]=(_Float16)f0.y; A.h[2]=(_Float16)f0.z; A.h[3]=(_Float16)f0.w;
    A.h[4]=(_Float16)f1.x; A.h[5]=(_Float16)f1.y; A.h[6]=(_Float16)f1.z; A.h[7]=(_Float16)f1.w;
    A.h[8]=(_Float16)f2.x; A.h[9]=(_Float16)f2.y; A.h[10]=(_Float16)f2.z; A.h[11]=(_Float16)f2.w;
    A.h[12]=(_Float16)f3.x; A.h[13]=(_Float16)f3.y; A.h[14]=(_Float16)f3.z; A.h[15]=(_Float16)f3.w;
    const _Float16* bp = Ct_i + ((size_t)e0) * 32 + m;
    FragH B0, B1;
    for (int q = 0; q < 16; q++) { int kq = (q < 8) ? (g * 8 + q) : (16 + g * 8 + q - 8); B0.h[q] = bp[kq * 32]; B1.h[q] = bp[kq * 32 + 16]; }
    ai0 = __builtin_amdgcn_wmma_f32_16x16x32_f16(false, A.v, false, B0.v, (short)0, ai0, false, false);
    ai1 = __builtin_amdgcn_wmma_f32_16x16x32_f16(false, A.v, false, B1.v, (short)0, ai1, false, false);
  }
  // D layout: N = lane%16 (=s), M = v + 8*(lane/16) (=t offset)
  int sN = lane & 15;
  #pragma unroll
  for (int v = 0; v < 8; v++) {
    int tM = t0 + v + g * 8;
    if (tM < TDATA) {
      if (sN < SUB) {
        syn_e[((size_t)(b * SUB + sN)) * TPAD + LEFTPAD + tM] = (_Float16)ae0[v];
        syn_i[((size_t)(b * SUB + sN)) * TPAD + LEFTPAD + tM] = (_Float16)ai0[v];
      }
      if (sN + 16 < SUB) {
        syn_e[((size_t)(b * SUB + sN + 16)) * TPAD + LEFTPAD + tM] = (_Float16)ae1[v];
        syn_i[((size_t)(b * SUB + sN + 16)) * TPAD + LEFTPAD + tM] = (_Float16)ai1[v];
      }
    }
  }
}

// ------- k4: causal conv as WMMA; async global->LDS window staging -------
// LDS layout (dwords): R_e[0..279], Rsh_e[280..559], R_i[560..839], Rsh_i[840..1119]
// Raw copy R starts one element BEFORE the window so the global source is dword
// aligned: win[k] = R[k+1]. Shifted copy Rsh[i] = R[i+1] = win[i] is built in LDS.
// Fragment B[k=lane][j] = win[off+j]: odd off reads R (dword (off+1)/2),
// even off reads Rsh (dword off/2).
__global__ void __launch_bounds__(256) k4_conv(const _Float16* __restrict__ syn_e,
                                               const _Float16* __restrict__ syn_i,
                                               const _Float16* __restrict__ frag_e,
                                               const _Float16* __restrict__ frag_i,
                                               const float* __restrict__ b1,
                                               const float* __restrict__ W2,
                                               float* __restrict__ sub_out) {
  __shared__ unsigned int sW[4 * 280];
  unsigned short* hW = (unsigned short*)sW;

  int wg = blockIdx.x;
  int b = wg / (SUB * NCHUNK);
  int rem = wg % (SUB * NCHUNK);
  int s = rem / NCHUNK;
  int chunkbase = (rem % NCHUNK) * 128;
  int gbase = LEFTPAD + chunkbase - (TNO - 1);        // >= 17, odd
  size_t rowbase = (size_t)(b * SUB + s) * TPAD;
  size_t base2 = rowbase + (size_t)(gbase - 1);       // even element offset -> 4B aligned

  const unsigned int* gse = (const unsigned int*)(((const unsigned short*)syn_e) + base2);
  const unsigned int* gsi = (const unsigned int*)(((const unsigned short*)syn_i) + base2);
  for (int d = threadIdx.x; d < 273; d += blockDim.x) {   // 546 halves = win[-1..544]
    async_copy_b32(gse + d, &sW[d]);
    async_copy_b32(gsi + d, &sW[560 + d]);
  }
  wait_async();
  __syncthreads();
  for (int i = threadIdx.x; i < 545; i += blockDim.x) {   // build shifted copies
    hW[560 + i]  = hW[i + 1];                             // Rsh_e
    hW[1680 + i] = hW[1120 + i + 1];                      // Rsh_i
  }
  __syncthreads();

  int wave = threadIdx.x >> 5;
  int lane = threadIdx.x & 31;
  int g = lane >> 4;
  int t0 = chunkbase + wave * 16;

  v8f acc = {};
  #pragma unroll 1
  for (int kstep = 0; kstep < NKSTEP; kstep++) {
    FragH A;
    const uint4* ap = (const uint4*)(frag_e + (((size_t)(s * NKSTEP + kstep)) * 32 + lane) * 16);
    A.q[0] = ap[0]; A.q[1] = ap[1];
    int off = wave * 16 + kstep * 32 + lane;
    const unsigned int* p = (off & 1) ? (sW + ((off + 1) >> 1)) : (sW + 280 + (off >> 1));
    FragH Bf;
    #pragma unroll
    for (int i = 0; i < 8; i++) Bf.u[i] = p[i];
    acc = __builtin_amdgcn_wmma_f32_16x16x32_f16(false, A.v, false, Bf.v, (short)0, acc, false, false);
  }
  #pragma unroll 1
  for (int kstep = 0; kstep < NKSTEP; kstep++) {
    FragH A;
    const uint4* ap = (const uint4*)(frag_i + (((size_t)(s * NKSTEP + kstep)) * 32 + lane) * 16);
    A.q[0] = ap[0]; A.q[1] = ap[1];
    int off = wave * 16 + kstep * 32 + lane;
    const unsigned int* p = (off & 1) ? (sW + 560 + ((off + 1) >> 1)) : (sW + 840 + (off >> 1));
    FragH Bf;
    #pragma unroll
    for (int i = 0; i < 8; i++) Bf.u[i] = p[i];
    acc = __builtin_amdgcn_wmma_f32_16x16x32_f16(false, A.v, false, Bf.v, (short)0, acc, false, false);
  }

  // epilogue: h = v + 8g across two lane halves; tanh + exp(W2) reduce over h
  float partial = 0.f;
  #pragma unroll
  for (int v = 0; v < 8; v++) {
    int h = v + g * 8;
    partial += expf(W2[s * HID + h]) * tanhf(acc[v] + b1[s * HID + h]);
  }
  float tot = partial + __shfl_xor(partial, 16, 32);
  int tN = t0 + (lane & 15);
  if (g == 0 && tN < TDATA) {
    sub_out[((size_t)(b * TDATA + tN)) * SUB + s] = tot;
  }
}

// ---------------- k5: final[b,t] = V_o + sum_s sub_out ----------------
__global__ void k5_final(const float* __restrict__ sub_out, const float* __restrict__ Vo,
                         float* __restrict__ fin) {
  int i = blockIdx.x * blockDim.x + threadIdx.x;
  if (i >= B_ * TDATA) return;
  float a = Vo[0];
  const float* p = sub_out + (size_t)i * SUB;
  #pragma unroll
  for (int s = 0; s < SUB; s++) a += p[s];
  fin[i] = a;
}

extern "C" void kernel_launch(void* const* d_in, const int* in_sizes, int n_in,
                              void* d_out, int out_size, void* d_ws, size_t ws_size,
                              hipStream_t stream) {
  const float* Se = (const float*)d_in[0];
  const float* Si = (const float*)d_in[1];
  const float* Escale = (const float*)d_in[2];
  const float* Iscale = (const float*)d_in[3];
  const float* We1 = (const float*)d_in[4];
  const float* Wi1 = (const float*)d_in[5];
  const float* W2 = (const float*)d_in[6];
  const float* b1 = (const float*)d_in[7];
  const float* CeR = (const float*)d_in[8];
  const float* CiR = (const float*)d_in[9];
  const float* Vo = (const float*)d_in[10];
  const float* temp = (const float*)d_in[11];

  char* ws = (char*)d_ws;
  _Float16* Ct_e = (_Float16*)(ws + OFF_CT_E);
  _Float16* Ct_i = (_Float16*)(ws + OFF_CT_I);
  _Float16* frag_e = (_Float16*)(ws + OFF_FRAG_E);
  _Float16* frag_i = (_Float16*)(ws + OFF_FRAG_I);
  _Float16* syn_e = (_Float16*)(ws + OFF_SYN_E);
  _Float16* syn_i = (_Float16*)(ws + OFF_SYN_I);

  float* out = (float*)d_out;
  float* o_final = out;               // 40000
  float* o_sub = out + 40000;         // 800000
  float* o_Ce = out + 840000;         // 40000
  float* o_Ci = out + 880000;         // 4000

  k0_zero<<<512, 256, 0, stream>>>((unsigned int*)ws, WS_BYTES / 4);
  k1_softmax<<<(ENO + INO + 255) / 256, 256, 0, stream>>>(CeR, CiR, Escale, Iscale, temp,
                                                          o_Ce, o_Ci, Ct_e, Ct_i);
  k3_kernels<<<(2 * SUB * HID * TKPAD + 255) / 256, 256, 0, stream>>>(We1, Wi1, frag_e, frag_i);
  k2_syn<<<626, 128, 0, stream>>>(Se, Si, Ct_e, Ct_i, syn_e, syn_i);
  k4_conv<<<B_ * SUB * NCHUNK, 256, 0, stream>>>(syn_e, syn_i, frag_e, frag_i, b1, W2, o_sub);
  k5_final<<<(B_ * TDATA + 255) / 256, 256, 0, stream>>>(o_sub, Vo, o_final);
}
